// GraphConvolution_52269751992444
// MI455X (gfx1250) — matrix operand order repaired and checked by reading, not compiled, not run on hardware
//
#include <hip/hip_runtime.h>
#include <hip/hip_bf16.h>
#include <cmath>

// ---------------------------------------------------------------------------
// Hyperbolic GCN layer for MI455X (gfx1250, wave32, WMMA).
//   mx = x @ W            -> bf16 WMMA (v_wmma_f32_16x16x32_bf16), f32 accum
//   mobius chain per row  -> wave-per-row, shfl_xor reductions
//   segment_sum scatter   -> atomicAdd f32 into L2-resident table
//   expmap0 + proj        -> wave-per-row
// ---------------------------------------------------------------------------

typedef __attribute__((ext_vector_type(16))) __bf16 bf16x16;
typedef __attribute__((ext_vector_type(8)))  __bf16 bf16x8;
typedef __attribute__((ext_vector_type(8)))  float  f32x8;

#define IN_DIM   256
#define OUT_DIM  64
#define RPB      32          // rows per block in GEMM
#define AST      264         // LDS row stride (bf16 elems), 16B-aligned
#define MIN_NORM 1e-15f
#define BALL_EPS 4e-3f

__device__ __forceinline__ float artanh_clip(float v) {
    v = fminf(fmaxf(v, -1.0f + 1e-7f), 1.0f - 1e-7f);
    return 0.5f * (log1pf(v) - log1pf(-v));
}

__device__ __forceinline__ float wred(float v) {
#pragma unroll
    for (int off = 16; off > 0; off >>= 1)
        v += __shfl_xor(v, off, 32);
    return v;
}

// ---------------------------------------------------------------------------
// Kernel 1: mx = x @ W via bf16 WMMA. Block = 256 thr = 8 waves = 2x4 tiles.
// ---------------------------------------------------------------------------
__global__ void hgcn_gemm_wmma(const float* __restrict__ x,
                               const float* __restrict__ w,
                               float* __restrict__ mx, int N)
{
    __shared__ __align__(16) __bf16 As[RPB][AST];       // x tile,   row-major
    __shared__ __align__(16) __bf16 Bt[OUT_DIM][AST];   // W^T tile: [col][k]

    const int tid  = threadIdx.x;
    const int row0 = blockIdx.x * RPB;

    // Stage weight [256][64] (row-major) transposed -> Bt[col][k] as bf16.
    for (int i = tid; i < IN_DIM * OUT_DIM; i += 256) {
        int k = i >> 6, col = i & 63;
        Bt[col][k] = (__bf16)w[i];
    }
    // Stage 32 rows of x as bf16 (float4 global loads, clamped row index).
    for (int i = tid; i < RPB * (IN_DIM / 4); i += 256) {
        int r  = i >> 6;            // 64 float4 per row
        int c4 = (i & 63) << 2;
        int gr = row0 + r; if (gr >= N) gr = N - 1;
        const float4 v = *(const float4*)(x + (size_t)gr * IN_DIM + c4);
        As[r][c4 + 0] = (__bf16)v.x; As[r][c4 + 1] = (__bf16)v.y;
        As[r][c4 + 2] = (__bf16)v.z; As[r][c4 + 3] = (__bf16)v.w;
    }
    __syncthreads();

    const int wave = tid >> 5, lane = tid & 31;
    const int rt = wave >> 2, ct = wave & 3;     // 2 row-tiles x 4 col-tiles
    const int l16 = lane & 15, half = lane >> 4;
    const __bf16* arow = &As[rt * 16 + l16][0];  // A: lane = M row
    const __bf16* bcol = &Bt[ct * 16 + l16][0];  // B: lane = N col
    const int ka = half * 8;                     // A half-lane K offset
    const int kb = half * 8;                     // B half-lane K offset

    f32x8 acc = {0.f, 0.f, 0.f, 0.f, 0.f, 0.f, 0.f, 0.f};
#pragma unroll
    for (int kk = 0; kk < IN_DIM / 32; ++kk) {
        const int base = kk * 32;
        // A 16x32 bf16: elems 0..7 -> K=base+ka+i ; 8..15 -> K=base+16+ka+i
        bf16x8 alo = *(const bf16x8*)(arow + base + ka);
        bf16x8 ahi = *(const bf16x8*)(arow + base + 16 + ka);
        // B 32x16 bf16: elems 0..7 -> K=base+kb+i ; 8..15 -> K=base+16+kb+i
        bf16x8 blo = *(const bf16x8*)(bcol + base + kb);
        bf16x8 bhi = *(const bf16x8*)(bcol + base + 16 + kb);
        bf16x16 a, b;
#pragma unroll
        for (int i = 0; i < 8; ++i) {
            a[i] = alo[i]; a[i + 8] = ahi[i];
            b[i] = blo[i]; b[i + 8] = bhi[i];
        }
        acc = __builtin_amdgcn_wmma_f32_16x16x32_bf16(
                  false, a, false, b, (short)0, acc, false, false);
    }

    // C/D layout: VGPR j -> M = j + 8*half, N = lane%16
#pragma unroll
    for (int j = 0; j < 8; ++j) {
        int gr = row0 + rt * 16 + j + half * 8;
        if (gr < N)
            mx[(size_t)gr * OUT_DIM + ct * 16 + l16] = acc[j];
    }
}

// ---------------------------------------------------------------------------
// Kernel 2: per-row mobius_matvec scale + mobius_add(bias) + logmap0.
// One wave per row; h is read (mx) and overwritten (logmap output) in place.
// ---------------------------------------------------------------------------
__global__ void hgcn_rowops(const float* __restrict__ x,
                            const float* __restrict__ bias,
                            const float* __restrict__ cptr,
                            float* __restrict__ h, int N)
{
    const int row  = blockIdx.x * 8 + (threadIdx.x >> 5);
    const int lane = threadIdx.x & 31;
    if (row >= N) return;                 // wave-uniform exit

    const float c  = *cptr;
    const float sc = sqrtf(c);

    // x_norm over 256 fp32
    float s = 0.f;
    const float* xr = x + (size_t)row * IN_DIM;
#pragma unroll
    for (int i = 0; i < IN_DIM / 32; ++i) {
        float v = xr[lane + i * 32];
        s += v * v;
    }
    const float x_norm = fmaxf(sqrtf(wred(s)), MIN_NORM);

    // mobius_matvec tail
    float m0 = h[(size_t)row * OUT_DIM + lane];
    float m1 = h[(size_t)row * OUT_DIM + lane + 32];
    const float ms = wred(m0 * m0 + m1 * m1);
    const float mx_norm = fmaxf(sqrtf(ms), MIN_NORM);
    const float t = tanhf(mx_norm / x_norm * artanh_clip(sc * x_norm));
    const float rscale = (ms == 0.f) ? 0.f : t / (mx_norm * sc);
    float r0 = m0 * rscale, r1 = m1 * rscale;

    // hyp_bias = proj(expmap0(bias, c), c)
    float b0 = bias[lane], b1 = bias[lane + 32];
    const float b_norm = fmaxf(sqrtf(wred(b0 * b0 + b1 * b1)), MIN_NORM);
    const float bt = tanhf(sc * b_norm) / (sc * b_norm);
    float hb0 = b0 * bt, hb1 = b1 * bt;
    const float hbn = fmaxf(sqrtf(wred(hb0 * hb0 + hb1 * hb1)), MIN_NORM);
    const float maxnorm = (1.0f - BALL_EPS) / sc;
    if (hbn > maxnorm) { float f = maxnorm / hbn; hb0 *= f; hb1 *= f; }

    // mobius_add(res, hyp_bias, c)
    const float x2 = wred(r0 * r0 + r1 * r1);
    const float y2 = wred(hb0 * hb0 + hb1 * hb1);
    const float xy = wred(r0 * hb0 + r1 * hb1);
    const float A = 1.0f + 2.0f * c * xy + c * y2;
    const float B = 1.0f - c * x2;
    const float den = fmaxf(1.0f + 2.0f * c * xy + c * c * x2 * y2, MIN_NORM);
    const float o0 = (A * r0 + B * hb0) / den;
    const float o1 = (A * r1 + B * hb1) / den;

    // logmap0
    const float pn = fmaxf(sqrtf(wred(o0 * o0 + o1 * o1)), MIN_NORM);
    const float lg = artanh_clip(sc * pn) / (sc * pn);
    h[(size_t)row * OUT_DIM + lane]      = o0 * lg;
    h[(size_t)row * OUT_DIM + lane + 32] = o1 * lg;
}

// ---------------------------------------------------------------------------
// Kernel 3: zero the aggregation buffer (graph-capture-safe init).
// ---------------------------------------------------------------------------
__global__ void hgcn_zero(float* __restrict__ p, long long n)
{
    long long i = (long long)blockIdx.x * blockDim.x + threadIdx.x;
    if (i < n) p[i] = 0.f;
}

// ---------------------------------------------------------------------------
// Kernel 4: SpMM scatter: agg[row[e]] += vals[e] * h[col[e]].  16 thr/edge.
// ---------------------------------------------------------------------------
__global__ void hgcn_scatter(const float* __restrict__ vals,
                             const int* __restrict__ rowi,
                             const int* __restrict__ coli,
                             const float* __restrict__ h,
                             float* __restrict__ agg, int E)
{
    long long t = (long long)blockIdx.x * blockDim.x + threadIdx.x;
    int e = (int)(t >> 4);
    if (e >= E) return;
    const int cg = ((int)t & 15) << 2;
    const float v = vals[e];
    const float4 hv = *(const float4*)(h + (size_t)coli[e] * OUT_DIM + cg);
    float* dst = agg + (size_t)rowi[e] * OUT_DIM + cg;
    atomicAdd(dst + 0, v * hv.x);
    atomicAdd(dst + 1, v * hv.y);
    atomicAdd(dst + 2, v * hv.z);
    atomicAdd(dst + 3, v * hv.w);
}

// ---------------------------------------------------------------------------
// Kernel 5: out = proj(expmap0(agg, c), c).  One wave per row.
// ---------------------------------------------------------------------------
__global__ void hgcn_finalize(const float* __restrict__ agg,
                              const float* __restrict__ cptr,
                              float* __restrict__ out, int N)
{
    const int row  = blockIdx.x * 8 + (threadIdx.x >> 5);
    const int lane = threadIdx.x & 31;
    if (row >= N) return;

    const float c  = *cptr;
    const float sc = sqrtf(c);

    float u0 = agg[(size_t)row * OUT_DIM + lane];
    float u1 = agg[(size_t)row * OUT_DIM + lane + 32];
    const float un = fmaxf(sqrtf(wred(u0 * u0 + u1 * u1)), MIN_NORM);
    const float em = tanhf(sc * un) / (sc * un);
    float e0 = u0 * em, e1 = u1 * em;

    const float en = fmaxf(sqrtf(wred(e0 * e0 + e1 * e1)), MIN_NORM);
    const float maxnorm = (1.0f - BALL_EPS) / sc;
    if (en > maxnorm) { float f = maxnorm / en; e0 *= f; e1 *= f; }

    out[(size_t)row * OUT_DIM + lane]      = e0;
    out[(size_t)row * OUT_DIM + lane + 32] = e1;
}

// ---------------------------------------------------------------------------
extern "C" void kernel_launch(void* const* d_in, const int* in_sizes, int n_in,
                              void* d_out, int out_size, void* d_ws, size_t ws_size,
                              hipStream_t stream)
{
    const float* x    = (const float*)d_in[0];
    const float* w    = (const float*)d_in[1];
    const float* bias = (const float*)d_in[2];
    const float* vals = (const float*)d_in[3];
    const float* cptr = (const float*)d_in[4];
    const int*   rowi = (const int*)d_in[5];
    const int*   coli = (const int*)d_in[6];
    float*       out  = (float*)d_out;

    const int N = in_sizes[0] / IN_DIM;   // 50000
    const int E = in_sizes[3];            // 800000

    float* h   = (float*)d_ws;                 // N*64 f32: mx -> logmap0 out
    float* agg = h + (size_t)N * OUT_DIM;      // N*64 f32: segment_sum accum

    hgcn_gemm_wmma<<<(N + RPB - 1) / RPB, 256, 0, stream>>>(x, w, h, N);
    hgcn_rowops  <<<(N + 7) / 8,          256, 0, stream>>>(x, bias, cptr, h, N);

    const long long nagg = (long long)N * OUT_DIM;
    hgcn_zero    <<<(unsigned)((nagg + 255) / 256), 256, 0, stream>>>(agg, nagg);

    const long long nscat = (long long)E * 16;
    hgcn_scatter <<<(unsigned)((nscat + 255) / 256), 256, 0, stream>>>(
        vals, rowi, coli, h, agg, E);

    hgcn_finalize<<<(N + 7) / 8, 256, 0, stream>>>(agg, cptr, out, N);
}
